// MosaicBlockLayer_81844896792922
// MI455X (gfx1250) — matrix operand order repaired and checked by reading, not compiled, not run on hardware
//
#include <hip/hip_runtime.h>
#include <hip/hip_bf16.h>
#include <math.h>

// ---------------------------------------------------------------------------
// MosaicBlockLayer for MI455X (gfx1250, wave32, WMMA)
// B=4 T=512 D=1024 K=16 CK=7 HID=2048 H=4 NB=128 NBITS=7 A=8 DK=64 DM=512
// ---------------------------------------------------------------------------

#define BB 4
#define TT 512
#define DD 1024
#define KK 16
#define CK 7
#define HID 2048
#define HH 4
#define NB 128
#define NBITS 7
#define AA 8
#define DKk 64
#define DM 512
#define NTOK (BB * TT)          // 2048
#define KD (KK * DD)            // 16384

typedef _Float16 h16;
typedef __attribute__((ext_vector_type(16))) _Float16 v16h;
typedef __attribute__((ext_vector_type(8)))  _Float16 v8h;
typedef __attribute__((ext_vector_type(8)))  float    v8f;

union V16H { v16h v; v8h h[2]; };

__device__ __forceinline__ float sigm(float x) { return 1.0f / (1.0f + __expf(-x)); }

__device__ __forceinline__ v8f wmma_f16(const V16H& a, const V16H& b, v8f c) {
    return __builtin_amdgcn_wmma_f32_16x16x32_f16(false, a.v, false, b.v,
                                                  (short)0, c, false, false);
}

// ---------------------------------------------------------------------------
// Weight convert + transpose: W (R x C) f32 row-major  ->  WT (C x R) f16
// ---------------------------------------------------------------------------
__global__ void cvt_transpose_f16(const float* __restrict__ W, h16* __restrict__ WT,
                                  int R, int C) {
    int idx = blockIdx.x * blockDim.x + threadIdx.x;
    if (idx >= R * C) return;
    int r = idx / C, c = idx % C;
    WT[(size_t)c * R + r] = (h16)W[idx];
}

// ---------------------------------------------------------------------------
// RMSNorm: per token row of D=1024, writes f32 and f16 copies
// ---------------------------------------------------------------------------
__global__ __launch_bounds__(256)
void rmsnorm_kernel(const float* __restrict__ x, float* __restrict__ u32o,
                    h16* __restrict__ u16o) {
    __shared__ float red[256];
    const int row = blockIdx.x, tid = threadIdx.x;
    const float* xr = x + (size_t)row * DD;
    float s = 0.0f;
    for (int i = tid; i < DD; i += 256) { float v = xr[i]; s += v * v; }
    red[tid] = s;
    __syncthreads();
    for (int st = 128; st > 0; st >>= 1) {
        if (tid < st) red[tid] += red[tid + st];
        __syncthreads();
    }
    const float scale = rsqrtf(red[0] / (float)DD + 1e-6f);
    for (int i = tid; i < DD; i += 256) {
        float v = xr[i] * scale;
        u32o[(size_t)row * DD + i] = v;
        u16o[(size_t)row * DD + i] = (h16)v;
    }
}

// ---------------------------------------------------------------------------
// WMMA GEMM, 32x32 register-blocked: each wave owns a 32(M) x 32(N) tile of C
// (4 accumulators), so each A/B fragment feeds two v_wmma_f32_16x16x32_f16 —
// 2 x b128 loads per WMMA instead of 4.
// A fragment (16-bit 16x32): lane l: row=l&15; halves 0..15 = K[kb..kb+15],
// kb=(l>>4)*16.
// B fragment: lane l: col=l&15; halves 0..15 = K[(l>>4)*16 .. +15] of BT row.
// ---------------------------------------------------------------------------
__global__ __launch_bounds__(32)
void wmma_gemm_kernel(const h16* __restrict__ A, const h16* __restrict__ BT,
                      const float* __restrict__ bias, float* __restrict__ C,
                      int M, int N, int K) {
    const int lane = threadIdx.x & 31;
    const int n0 = blockIdx.x * 32;
    const int m0 = blockIdx.y * 32;

    const int r15 = lane & 15;
    const int kbA = (lane >> 4) * 16;
    const int kbB = (lane >> 4) * 16;

    const h16* pa0 = A  + (size_t)(m0 + r15) * K + kbA;
    const h16* pa1 = A  + (size_t)(m0 + 16 + r15) * K + kbA;
    const h16* pb0 = BT + (size_t)(n0 + r15) * K + kbB;
    const h16* pb1 = BT + (size_t)(n0 + 16 + r15) * K + kbB;

    v8f acc00 = {}, acc01 = {}, acc10 = {}, acc11 = {};
    for (int k0 = 0; k0 < K; k0 += 32) {
        V16H a0, a1, b0, b1;
        a0.h[0] = *(const v8h*)(pa0 + k0);
        a0.h[1] = *(const v8h*)(pa0 + k0 + 8);
        a1.h[0] = *(const v8h*)(pa1 + k0);
        a1.h[1] = *(const v8h*)(pa1 + k0 + 8);
        b0.h[0] = *(const v8h*)(pb0 + k0);
        b0.h[1] = *(const v8h*)(pb0 + k0 + 8);
        b1.h[0] = *(const v8h*)(pb1 + k0);
        b1.h[1] = *(const v8h*)(pb1 + k0 + 8);
        __builtin_prefetch(pa0 + k0 + 64, 0, 1);   // global_prefetch_b8
        __builtin_prefetch(pa1 + k0 + 64, 0, 1);
        __builtin_prefetch(pb0 + k0 + 64, 0, 1);
        __builtin_prefetch(pb1 + k0 + 64, 0, 1);
        acc00 = wmma_f16(a0, b0, acc00);
        acc01 = wmma_f16(a0, b1, acc01);
        acc10 = wmma_f16(a1, b0, acc10);
        acc11 = wmma_f16(a1, b1, acc11);
    }

    const int rbase = (lane >> 4) * 8;
    const float bv0 = bias ? bias[n0 + r15] : 0.0f;
    const float bv1 = bias ? bias[n0 + 16 + r15] : 0.0f;
#pragma unroll
    for (int r = 0; r < 8; ++r) {
        C[(size_t)(m0 + rbase + r) * N + n0 + r15]           = acc00[r] + bv0;
        C[(size_t)(m0 + rbase + r) * N + n0 + 16 + r15]      = acc01[r] + bv1;
        C[(size_t)(m0 + 16 + rbase + r) * N + n0 + r15]      = acc10[r] + bv0;
        C[(size_t)(m0 + 16 + rbase + r) * N + n0 + 16 + r15] = acc11[r] + bv1;
    }
}

// ---------------------------------------------------------------------------
// Depthwise causal conv (CK=7) * SiLU(gate_lin), result as f16 for MLP GEMM
// ---------------------------------------------------------------------------
__global__ void dwconv_gate_kernel(const float* __restrict__ u,
                                   const float* __restrict__ wconv,
                                   const float* __restrict__ gate_lin,
                                   h16* __restrict__ mixed16) {
    int idx = blockIdx.x * blockDim.x + threadIdx.x;
    if (idx >= NTOK * DD) return;
    int d = idx % DD, bt = idx / DD;
    int t = bt % TT, b = bt / TT;
    float acc = 0.0f;
#pragma unroll
    for (int j = 0; j < CK; ++j) {
        int tt = t - (CK - 1) + j;
        if (tt >= 0)
            acc += u[((size_t)(b * TT + tt)) * DD + d] * wconv[d * CK + j];
    }
    float g = gate_lin[idx];
    mixed16[idx] = (h16)(acc * (g * sigm(g)));
}

// ---------------------------------------------------------------------------
// GELU (tanh approx, matches jax.nn.gelu default) -> f16
// ---------------------------------------------------------------------------
__global__ void gelu_to_f16(const float* __restrict__ in, h16* __restrict__ out, int n) {
    int i = blockIdx.x * blockDim.x + threadIdx.x;
    if (i >= n) return;
    float x = in[i];
    float t = tanhf(0.7978845608028654f * (x + 0.044715f * x * x * x));
    out[i] = (h16)(0.5f * x * (1.0f + t));
}

// ---------------------------------------------------------------------------
// Decay-gated state-bank scan: s_t = a*s_{t-1} + (1-a)*v_t, per (b,k,d).
// Writes s_all as f16 (feeds the KD->D WMMA GEMM).
// ---------------------------------------------------------------------------
__global__ __launch_bounds__(256)
void state_scan_kernel(const float* __restrict__ v,
                       const float* __restrict__ decay_logit,
                       h16* __restrict__ s16) {
    int gid = blockIdx.x * 256 + threadIdx.x;
    if (gid >= BB * KD) return;
    int b = gid / KD, kd = gid % KD, k = kd / DD;
    float a  = sigm(decay_logit[k]);
    float om = 1.0f - a;
    float s = 0.0f;
    size_t base = (size_t)b * TT * KD + kd;
    for (int t = 0; t < TT; ++t) {
        s = a * s + om * v[base + (size_t)t * KD];
        s16[base + (size_t)t * KD] = (h16)s;
    }
}

// ---------------------------------------------------------------------------
// Routing: 59 dot products per token (2*H*NBITS bit planes + 3 scalar gates).
// The 4 KB u-row is staged into LDS via GLOBAL_LOAD_ASYNC_TO_LDS_B128
// (each of 256 lanes moves one 16-byte chunk, tracked by ASYNCcnt).
// ---------------------------------------------------------------------------
#define NOUT (2 * HH * NBITS + 3)   // 59
__global__ __launch_bounds__(256)
void routing_kernel(const float* __restrict__ u,
                    const float* __restrict__ w_rbits, const float* __restrict__ w_wbits,
                    const float* __restrict__ w_wg, const float* __restrict__ b_wg,
                    const float* __restrict__ w_gl, const float* __restrict__ b_gl,
                    const float* __restrict__ w_gm, const float* __restrict__ b_gm,
                    int* __restrict__ rbuck, int* __restrict__ wbuck,
                    float* __restrict__ gw, float* __restrict__ gl,
                    float* __restrict__ gm) {
    __shared__ float us[DD];
    __shared__ float outs[NOUT];
    const int tok = blockIdx.x, tid = threadIdx.x;
    const float* ur = u + (size_t)tok * DD;

    // Async stage: 256 lanes x 16B = 4 KB row into LDS.
    // Flat LDS addresses map via addr[31:0] (ISA 10.2), so the low 32 bits of
    // the generic shared pointer are the per-lane LDS byte address.
    {
        unsigned lds_addr = (unsigned)(size_t)us + (unsigned)(tid * 16);
        unsigned long long gaddr =
            (unsigned long long)(const void*)ur + (unsigned long long)tid * 16ull;
        asm volatile("global_load_async_to_lds_b128 %0, %1, off"
                     :: "v"(lds_addr), "v"(gaddr) : "memory");
        asm volatile("s_wait_asynccnt 0x0" ::: "memory");
    }
    __syncthreads();

    const int wave = tid >> 5, lane = tid & 31;
    for (int o = wave; o < NOUT; o += 8) {
        float p = 0.0f;
        if (o < HH * NBITS) {
            for (int i = lane; i < DD; i += 32) p += us[i] * w_rbits[i * (HH * NBITS) + o];
        } else if (o < 2 * HH * NBITS) {
            int j = o - HH * NBITS;
            for (int i = lane; i < DD; i += 32) p += us[i] * w_wbits[i * (HH * NBITS) + j];
        } else if (o == 56) {
            for (int i = lane; i < DD; i += 32) p += us[i] * w_wg[i];
        } else if (o == 57) {
            for (int i = lane; i < DD; i += 32) p += us[i] * w_gl[i];
        } else {
            for (int i = lane; i < DD; i += 32) p += us[i] * w_gm[i];
        }
        for (int m = 16; m > 0; m >>= 1) p += __shfl_xor(p, m, 32);
        if (lane == 0) outs[o] = p;
    }
    __syncthreads();

    if (tid < HH) {
        int rb = 0, wb = 0;
#pragma unroll
        for (int bit = 0; bit < NBITS; ++bit) {
            if (outs[tid * NBITS + bit] > 0.0f) rb |= (1 << bit);
            if (outs[HH * NBITS + tid * NBITS + bit] > 0.0f) wb |= (1 << bit);
        }
        rbuck[tok * HH + tid] = rb;
        wbuck[tok * HH + tid] = wb;
    } else if (tid == 4) gw[tok] = sigm(outs[56] + b_wg[0]);
    else if   (tid == 5) gl[tok] = sigm(outs[57] + b_gl[0]);
    else if   (tid == 6) gm[tok] = sigm(outs[58] + b_gm[0]);
}

// ---------------------------------------------------------------------------
// Memory state init (graph-replay safe: re-initializes every call)
// ---------------------------------------------------------------------------
__global__ void mem_init_kernel(float* __restrict__ mem_k, float* __restrict__ mem_v,
                                int* __restrict__ mem_last) {
    int i = blockIdx.x * 256 + threadIdx.x;
    if (i < BB * HH * NB * AA * DKk) mem_k[i] = 0.0f;
    if (i < BB * HH * NB * AA * DM)  mem_v[i] = 0.0f;
    if (i < BB * HH * NB * AA)       mem_last[i] = -1;
}

// ---------------------------------------------------------------------------
// Hashed associative memory scan. One workgroup (256 thr) per batch; state in
// global scratch (~9 MB/batch exceeds 320 KB LDS). Per t: scores -> masked
// softmax -> read (mean over heads) -> LRU write.
// ---------------------------------------------------------------------------
__global__ __launch_bounds__(256)
void mem_scan_kernel(const float* __restrict__ q, const float* __restrict__ kkey,
                     const float* __restrict__ val,
                     const int* __restrict__ rbuck, const int* __restrict__ wbuck,
                     const float* __restrict__ gw,
                     float* __restrict__ mem_k, float* __restrict__ mem_v,
                     int* __restrict__ mem_last, h16* __restrict__ reads16) {
    const int b = blockIdx.x, tid = threadIdx.x;
    __shared__ float sc[HH * AA];
    __shared__ float at[HH * AA];
    __shared__ int ssel[HH];
    __shared__ int srb[HH], swb[HH];
    const float inv_sqrt_dk = 0.125f;  // 1/sqrt(64)

    for (int t = 0; t < TT; ++t) {
        const int tok = b * TT + t;
        if (tid < HH) { srb[tid] = rbuck[tok * HH + tid]; swb[tid] = wbuck[tok * HH + tid]; }
        __syncthreads();

        // ---- scores: 32 (h,a) pairs, 8 threads each ----
        {
            const int p = tid >> 3, l8 = tid & 7;
            const int h = p >> 3, a = p & 7;
            const int rb = srb[h];
            const float* qh  = q + (size_t)tok * (HH * DKk) + h * DKk;
            const float* skp = mem_k + ((((size_t)b * HH + h) * NB + rb) * AA + a) * DKk;
            float pp = 0.0f;
            for (int i = l8; i < DKk; i += 8) pp += qh[i] * skp[i];
            pp += __shfl_xor(pp, 4, 8);
            pp += __shfl_xor(pp, 2, 8);
            pp += __shfl_xor(pp, 1, 8);
            if (l8 == 0) sc[p] = pp * inv_sqrt_dk;
        }
        __syncthreads();

        // ---- masked softmax per head + LRU slot selection ----
        if (tid < HH) {
            const int h = tid, rb = srb[h];
            const int* ml = mem_last + (((size_t)b * HH + h) * NB + rb) * AA;
            float s[AA], mx = -1e30f;
#pragma unroll
            for (int a = 0; a < AA; ++a) {
                s[a] = (ml[a] >= 0) ? sc[h * AA + a] : -1e9f;
                mx = fmaxf(mx, s[a]);
            }
            float sum = 0.0f;
#pragma unroll
            for (int a = 0; a < AA; ++a) { s[a] = __expf(s[a] - mx); sum += s[a]; }
            const float inv = 1.0f / sum;
#pragma unroll
            for (int a = 0; a < AA; ++a) at[h * AA + a] = s[a] * inv;

            const int wb = swb[h];
            const int* mlw = mem_last + (((size_t)b * HH + h) * NB + wb) * AA;
            int best = 0, bv = mlw[0];
#pragma unroll
            for (int a = 1; a < AA; ++a) if (mlw[a] < bv) { bv = mlw[a]; best = a; }
            ssel[h] = best;
        }
        __syncthreads();

        // ---- read: mean over heads of attn-weighted slot values ----
        for (int dm = tid; dm < DM; dm += 256) {
            float acc = 0.0f;
            for (int h = 0; h < HH; ++h) {
                const int rb = srb[h];
                const float* svp = mem_v + ((((size_t)b * HH + h) * NB + rb) * AA) * DM + dm;
#pragma unroll
                for (int a = 0; a < AA; ++a) acc += at[h * AA + a] * svp[(size_t)a * DM];
            }
            reads16[(size_t)tok * DM + dm] = (h16)(acc * (1.0f / HH));
        }
        __syncthreads();

        // ---- write: replace LRU slot in write bucket ----
        {
            const int h = tid >> 6, d = tid & 63;   // 256 = 4*64
            const int wb = swb[h];
            mem_k[((((size_t)b * HH + h) * NB + wb) * AA + ssel[h]) * DKk + d] =
                kkey[(size_t)tok * (HH * DKk) + h * DKk + d];
        }
        const float gt = gw[tok];
        for (int i = tid; i < HH * DM; i += 256) {
            const int h = i >> 9, d = i & (DM - 1);
            const int wb = swb[h];
            mem_v[((((size_t)b * HH + h) * NB + wb) * AA + ssel[h]) * DM + d] =
                gt * val[(size_t)tok * DM + d];
        }
        if (tid < HH) {
            const int wb = swb[tid];
            mem_last[(((size_t)b * HH + tid) * NB + wb) * AA + ssel[tid]] = t;
        }
        __syncthreads();
    }
}

// ---------------------------------------------------------------------------
// Final combine: out = x + local + sigmoid(gl)*long + sigmoid(gm)*mem_out
// (gl/gm already sigmoided in routing kernel)
// ---------------------------------------------------------------------------
__global__ void combine_kernel(const float* __restrict__ x, const float* __restrict__ localb,
                               const float* __restrict__ longb, const float* __restrict__ memo,
                               const float* __restrict__ gl, const float* __restrict__ gm,
                               float* __restrict__ out) {
    int idx = blockIdx.x * blockDim.x + threadIdx.x;
    if (idx >= NTOK * DD) return;
    int tok = idx / DD;
    out[idx] = x[idx] + localb[idx] + gl[tok] * longb[idx] + gm[tok] * memo[idx];
}

// ---------------------------------------------------------------------------
// Host launcher
// ---------------------------------------------------------------------------
extern "C" void kernel_launch(void* const* d_in, const int* in_sizes, int n_in,
                              void* d_out, int out_size, void* d_ws, size_t ws_size,
                              hipStream_t stream) {
    (void)in_sizes; (void)n_in; (void)out_size; (void)ws_size;

    const float* x       = (const float*)d_in[0];
    const float* w_conv  = (const float*)d_in[1];
    const float* w_gate  = (const float*)d_in[2];
    const float* b_gate  = (const float*)d_in[3];
    const float* w_up    = (const float*)d_in[4];
    const float* b_up    = (const float*)d_in[5];
    const float* w_down  = (const float*)d_in[6];
    const float* b_down  = (const float*)d_in[7];
    const float* w_in    = (const float*)d_in[8];
    const float* w_out   = (const float*)d_in[9];
    const float* decay   = (const float*)d_in[10];
    const float* wq      = (const float*)d_in[11];
    const float* wk      = (const float*)d_in[12];
    const float* wv      = (const float*)d_in[13];
    const float* wo      = (const float*)d_in[14];
    const float* w_rbits = (const float*)d_in[15];
    const float* w_wbits = (const float*)d_in[16];
    const float* w_wg    = (const float*)d_in[17];
    const float* b_wg    = (const float*)d_in[18];
    const float* w_gl    = (const float*)d_in[19];
    const float* b_gl    = (const float*)d_in[20];
    const float* w_gm    = (const float*)d_in[21];
    const float* b_gm    = (const float*)d_in[22];
    float* out = (float*)d_out;

    // ---- workspace carve-up ----
    char* ws = (char*)d_ws;
    size_t off = 0;
    auto take = [&](size_t bytes) -> void* {
        void* p = ws + off;
        off += (bytes + 255) & ~(size_t)255;
        return p;
    };
    float* u32     = (float*)take((size_t)NTOK * DD * 4);
    h16*   u16     = (h16*)  take((size_t)NTOK * DD * 2);
    h16*   wgT     = (h16*)  take((size_t)DD * DD * 2);
    h16*   wupT    = (h16*)  take((size_t)HID * DD * 2);
    h16*   wdnT    = (h16*)  take((size_t)DD * HID * 2);
    h16*   winT    = (h16*)  take((size_t)KD * DD * 2);
    h16*   woutT   = (h16*)  take((size_t)DD * KD * 2);
    h16*   wqT     = (h16*)  take((size_t)(HH * DKk) * DD * 2);
    h16*   wkT     = (h16*)  take((size_t)(HH * DKk) * DD * 2);
    h16*   wvT     = (h16*)  take((size_t)DM * DD * 2);
    h16*   woT     = (h16*)  take((size_t)DD * DM * 2);
    float* gatelin = (float*)take((size_t)NTOK * DD * 4);
    h16*   mixed16 = (h16*)  take((size_t)NTOK * DD * 2);
    float* hid     = (float*)take((size_t)NTOK * HID * 4);
    h16*   hid16   = (h16*)  take((size_t)NTOK * HID * 2);
    float* localb  = (float*)take((size_t)NTOK * DD * 4);
    float* vbuf    = (float*)take((size_t)NTOK * KD * 4);
    h16*   s16     = (h16*)  take((size_t)NTOK * KD * 2);
    float* longb   = (float*)take((size_t)NTOK * DD * 4);
    float* qbuf    = (float*)take((size_t)NTOK * HH * DKk * 4);
    float* kbuf    = (float*)take((size_t)NTOK * HH * DKk * 4);
    float* valbuf  = (float*)take((size_t)NTOK * DM * 4);
    int*   rbuckb  = (int*)  take((size_t)NTOK * HH * 4);
    int*   wbuckb  = (int*)  take((size_t)NTOK * HH * 4);
    float* gwb     = (float*)take((size_t)NTOK * 4);
    float* glb     = (float*)take((size_t)NTOK * 4);
    float* gmb     = (float*)take((size_t)NTOK * 4);
    float* memk    = (float*)take((size_t)BB * HH * NB * AA * DKk * 4);
    float* memv    = (float*)take((size_t)BB * HH * NB * AA * DM * 4);
    int*   memlast = (int*)  take((size_t)BB * HH * NB * AA * 4);
    h16*   reads16 = (h16*)  take((size_t)NTOK * DM * 2);
    float* memout  = (float*)take((size_t)NTOK * DD * 4);

    auto cvt = [&](const float* W, h16* WT, int R, int C) {
        int n = R * C;
        cvt_transpose_f16<<<(n + 255) / 256, 256, 0, stream>>>(W, WT, R, C);
    };
    auto gemm = [&](const h16* A, const h16* BT, const float* bias, float* C,
                    int M, int N, int Kdim) {
        dim3 g(N / 32, M / 32);
        wmma_gemm_kernel<<<g, 32, 0, stream>>>(A, BT, bias, C, M, N, Kdim);
    };

    // 1) weights -> f16 transposed (N x K)
    cvt(w_gate, wgT,  DD, DD);
    cvt(w_up,   wupT, DD, HID);
    cvt(w_down, wdnT, HID, DD);
    cvt(w_in,   winT, DD, KD);
    cvt(w_out,  woutT, KD, DD);
    cvt(wq,     wqT,  DD, HH * DKk);
    cvt(wk,     wkT,  DD, HH * DKk);
    cvt(wv,     wvT,  DD, DM);
    cvt(wo,     woT,  DM, DD);

    // 2) rmsnorm
    rmsnorm_kernel<<<NTOK, 256, 0, stream>>>(x, u32, u16);

    // 3) local mixer path
    gemm(u16, wgT, b_gate, gatelin, NTOK, DD, DD);
    {
        int n = NTOK * DD;
        dwconv_gate_kernel<<<(n + 255) / 256, 256, 0, stream>>>(u32, w_conv, gatelin, mixed16);
    }
    gemm(mixed16, wupT, b_up, hid, NTOK, HID, DD);
    {
        int n = NTOK * HID;
        gelu_to_f16<<<(n + 255) / 256, 256, 0, stream>>>(hid, hid16, n);
    }
    gemm(hid16, wdnT, b_down, localb, NTOK, DD, HID);

    // 4) decay-gated state bank
    gemm(u16, winT, nullptr, vbuf, NTOK, KD, DD);
    state_scan_kernel<<<(BB * KD + 255) / 256, 256, 0, stream>>>(vbuf, decay, s16);
    gemm(s16, woutT, nullptr, longb, NTOK, DD, KD);

    // 5) hashed associative memory
    gemm(u16, wqT, nullptr, qbuf, NTOK, HH * DKk, DD);
    gemm(u16, wkT, nullptr, kbuf, NTOK, HH * DKk, DD);
    gemm(u16, wvT, nullptr, valbuf, NTOK, DM, DD);
    routing_kernel<<<NTOK, 256, 0, stream>>>(u32, w_rbits, w_wbits, w_wg, b_wg,
                                             w_gl, b_gl, w_gm, b_gm,
                                             rbuckb, wbuckb, gwb, glb, gmb);
    {
        int n = BB * HH * NB * AA * DM;  // largest state array
        mem_init_kernel<<<(n + 255) / 256, 256, 0, stream>>>(memk, memv, memlast);
    }
    mem_scan_kernel<<<BB, 256, 0, stream>>>(qbuf, kbuf, valbuf, rbuckb, wbuckb, gwb,
                                            memk, memv, memlast, reads16);
    gemm(reads16, woT, nullptr, memout, NTOK, DD, DM);

    // 6) combine
    {
        int n = NTOK * DD;
        combine_kernel<<<(n + 255) / 256, 256, 0, stream>>>(x, localb, longb, memout,
                                                            glb, gmb, out);
    }
}
